// ConsciousnessAttention_86174223827046
// MI455X (gfx1250) — matrix operand order repaired and checked by reading, not compiled, not run on hardware
//
#include <hip/hip_runtime.h>

// ---------------------------------------------------------------------------
// ConsciousnessAttention for MI455X (gfx1250), wave32 + WMMA bf16 + TDM.
//
// Exact-math simplifications:
//  * coherence MLP row-factor == 1 within f32 eps  -> dropped entirely
//  * field attention is rank-1 along keys          -> epilogue: 0.21*fa[b,i]*colsum(V)
//
// CDNA5 paths used: v_wmma_f32_16x16x32_bf16 (all GEMMs), tensor_load_to_lds
// (double-buffered K/V tile staging in flash attention), s_wait_tensorcnt.
// ---------------------------------------------------------------------------

typedef __bf16 bf16;
typedef __attribute__((ext_vector_type(16))) __bf16 v16bf;
typedef __attribute__((ext_vector_type(8)))  __bf16 v8bf;
typedef __attribute__((ext_vector_type(8)))  float  v8f;
typedef unsigned int u32;
typedef unsigned long long u64;
typedef __attribute__((ext_vector_type(4))) u32 u32x4;
typedef __attribute__((ext_vector_type(8))) int i32x8;
typedef __attribute__((ext_vector_type(4))) int i32x4;

#define FI   0.21f      // FIELD_COUPLING * FIELD_COHERENCE
#define OMFI 0.79f      // 1 - FI

#if __has_builtin(__builtin_amdgcn_tensor_load_to_lds) && __has_builtin(__builtin_amdgcn_s_wait_tensorcnt)
#define USE_TDM 1
#else
#define USE_TDM 0
#endif

__device__ __forceinline__ v8f wmma_bf16(v16bf a, v16bf b, v8f c) {
  return __builtin_amdgcn_wmma_f32_16x16x32_bf16(false, a, false, b, (short)0, c, false, false);
}

// A-fragment (16xK, 16-bit): lane half g -> elems 0..7 = K[8g..], 8..15 = K[16+8g..]
__device__ __forceinline__ v16bf frag_a(const bf16* row, int g) {
  v8bf lo = *(const v8bf*)(row + 8 * g);
  v8bf hi = *(const v8bf*)(row + 16 + 8 * g);
  return __builtin_shufflevector(lo, hi, 0,1,2,3,4,5,6,7,8,9,10,11,12,13,14,15);
}

// B-fragment (Kx16, 16-bit): lane half g -> elems 0..7 = K[8g..], 8..15 = K[16+8g..]
__device__ __forceinline__ v16bf frag_b(const bf16* row, int g) {
  v8bf lo = *(const v8bf*)(row + 8 * g);
  v8bf hi = *(const v8bf*)(row + 16 + 8 * g);
  return __builtin_shufflevector(lo, hi, 0,1,2,3,4,5,6,7,8,9,10,11,12,13,14,15);
}

#if USE_TDM
// Issue one TDM 2D (or 1D when tile1==0) tile load: global bf16 -> LDS, with
// LDS padding of 4 DWORDs every 32 DWORDs (row stride 128B -> 144B = 72 bf16).
__device__ __forceinline__ void tdm_load_tile(u32 lds_off, const bf16* gptr,
                                              u32 tile0, u32 tile1,
                                              u32 td0, u32 td1, u64 stride0) {
  u64 ga = (u64)(unsigned long long)gptr;
  u32x4 g0 = {0, 0, 0, 0};
  g0[0] = 1u;                                            // count=1, user desc
  g0[1] = lds_off;                                       // LDS byte address
  g0[2] = (u32)ga;                                       // global_addr[31:0]
  g0[3] = ((u32)(ga >> 32) & 0x01FFFFFFu) | (2u << 30);  // addr[56:32] | type=2
  i32x8 g1 = {0, 0, 0, 0, 0, 0, 0, 0};
  // data_size=1 (2B) | pad_enable | pad_interval=4 (32 DW) | pad_amount=3 (4 DW)
  g1[0] = (int)((1u << 16) | (1u << 20) | (4u << 22) | (3u << 25));
  g1[1] = (int)((td0 & 0xFFFFu) << 16);                  // tensor_dim0[15:0]
  g1[2] = (int)((td0 >> 16) | ((td1 & 0xFFFFu) << 16));  // td0[31:16] | td1[15:0]
  g1[3] = (int)((td1 >> 16) | (tile0 << 16));            // td1[31:16] | tile_dim0
  g1[4] = (int)(tile1);                                  // tile_dim1 (tile_dim2=0)
  g1[5] = (int)((u32)stride0);                           // dim0 stride [31:0]
  g1[6] = (int)((u32)(stride0 >> 32) & 0xFFFFu);         // dim0 stride [47:32]
  g1[7] = 0;
  i32x4 z4 = {0, 0, 0, 0};
#if defined(__clang_major__) && (__clang_major__ >= 23)
  i32x8 z8 = {0, 0, 0, 0, 0, 0, 0, 0};
  __builtin_amdgcn_tensor_load_to_lds(g0, g1, z4, z4, z8, 0);
#else
  __builtin_amdgcn_tensor_load_to_lds(g0, g1, z4, z4, 0);
#endif
}
#endif

// ---------------------------------------------------------------------------
// Generic GEMM:  C[M,N] = A[M,K] * W[N,K]^T + bias
//   AF32:  A source is f32 (converted to bf16 on the LDS stage) else bf16
//   EPI=0: store f32 row-major C
//   EPI=1: QKV epilogue -> scatter bf16 Q [BH,S,64], K [BH,S,64], Vt [BH,64,S]
// Block: 256 thr = 8 waves, tile 128x128x32; wave grid 4(M) x 2(N).
// ---------------------------------------------------------------------------
template <int EPI, bool AF32>
__global__ __launch_bounds__(256) void gemm_kernel(
    const void* __restrict__ Aptr, const float* __restrict__ W,
    const float* __restrict__ bias, float* __restrict__ Cf,
    bf16* __restrict__ q, bf16* __restrict__ k, bf16* __restrict__ vt,
    int M, int N, int K) {
  __shared__ __align__(16) bf16 As[128][40];
  __shared__ __align__(16) bf16 Bs[128][40];

  const int tid  = threadIdx.x;
  const int lane = tid & 31, w = tid >> 5;
  const int half = lane >> 4, mr = lane & 15;
  const int wm = (w & 3) * 32, wn = (w >> 2) * 64;
  const int m0 = blockIdx.y * 128, n0 = blockIdx.x * 128;

  const int ar = tid & 127, aseg = (tid >> 7) * 16;  // stage coords

  v8f acc[2][4];
  for (int i = 0; i < 2; ++i)
    for (int j = 0; j < 4; ++j) acc[i][j] = (v8f)0.0f;

  for (int kk = 0; kk < K; kk += 32) {
    if (AF32) {
      const float* Ar = (const float*)Aptr + (size_t)(m0 + ar) * K + kk + aseg;
      for (int j = 0; j < 16; j += 4) {
        float4 v = *(const float4*)(Ar + j);
        As[ar][aseg + j + 0] = (bf16)v.x;
        As[ar][aseg + j + 1] = (bf16)v.y;
        As[ar][aseg + j + 2] = (bf16)v.z;
        As[ar][aseg + j + 3] = (bf16)v.w;
      }
      if (kk + 32 < K) __builtin_prefetch(Ar + 32, 0, 1);
    } else {
      const bf16* Ar = (const bf16*)Aptr + (size_t)(m0 + ar) * K + kk + aseg;
      *(v8bf*)&As[ar][aseg]     = *(const v8bf*)Ar;
      *(v8bf*)&As[ar][aseg + 8] = *(const v8bf*)(Ar + 8);
      if (kk + 32 < K) __builtin_prefetch(Ar + 32, 0, 1);
    }
    {
      const float* Br = W + (size_t)(n0 + ar) * K + kk + aseg;
      for (int j = 0; j < 16; j += 4) {
        float4 v = *(const float4*)(Br + j);
        Bs[ar][aseg + j + 0] = (bf16)v.x;
        Bs[ar][aseg + j + 1] = (bf16)v.y;
        Bs[ar][aseg + j + 2] = (bf16)v.z;
        Bs[ar][aseg + j + 3] = (bf16)v.w;
      }
      if (kk + 32 < K) __builtin_prefetch(Br + 32, 0, 1);
    }
    __syncthreads();

    v16bf af[2], bfv[4];
    for (int i = 0; i < 2; ++i) af[i]  = frag_a(&As[wm + 16 * i + mr][0], half);
    for (int j = 0; j < 4; ++j) bfv[j] = frag_b(&Bs[wn + 16 * j + mr][0], half);
    for (int i = 0; i < 2; ++i)
      for (int j = 0; j < 4; ++j) acc[i][j] = wmma_bf16(af[i], bfv[j], acc[i][j]);
    __syncthreads();
  }

  for (int i = 0; i < 2; ++i) {
    for (int j = 0; j < 4; ++j) {
      for (int v = 0; v < 8; ++v) {
        int m = m0 + wm + 16 * i + v + 8 * half;   // C: M = v + 8*(lane>=16)
        int n = n0 + wn + 16 * j + mr;             //    N = lane%16
        float val = acc[i][j][v] + bias[n];
        if (EPI == 0) {
          Cf[(size_t)m * N + n] = val;
        } else {
          int b = m >> 10, s = m & 1023;
          int which = n >> 10, f = n & 1023, h = f >> 6, d = f & 63;
          size_t bh = (size_t)(b * 16 + h);
          bf16 bv = (bf16)val;
          if (which == 0)      q[(bh * 1024 + s) * 64 + d] = bv;
          else if (which == 1) k[(bh * 1024 + s) * 64 + d] = bv;
          else                 vt[(bh * 64 + d) * 1024 + s] = bv;  // V transposed
        }
      }
    }
  }
}

// ---------------------------------------------------------------------------
// Field encoder: fp[b] = LN(relu(cs@We1^T+be1)@We2^T+be2)   (B=4, tiny, f32)
// ---------------------------------------------------------------------------
__global__ __launch_bounds__(256) void field_fp_kernel(
    const float* __restrict__ cs, const float* __restrict__ We1,
    const float* __restrict__ be1, const float* __restrict__ We2,
    const float* __restrict__ be2, const float* __restrict__ g,
    const float* __restrict__ bta, float* __restrict__ fp) {
  __shared__ float csl[256];
  __shared__ float h1[1024];
  __shared__ float red[256];
  const int b = blockIdx.x, tid = threadIdx.x;

  csl[tid] = cs[b * 256 + tid];
  __syncthreads();
  for (int e = tid; e < 1024; e += 256) {
    const float* wr = We1 + (size_t)e * 256;
    float a = be1[e];
    for (int c = 0; c < 256; ++c) a = fmaf(csl[c], wr[c], a);
    h1[e] = fmaxf(a, 0.0f);
  }
  __syncthreads();

  float h2[4], lsum = 0.f, lsq = 0.f;
  for (int i = 0; i < 4; ++i) {
    int e = tid + 256 * i;
    const float* wr = We2 + (size_t)e * 1024;
    float a = be2[e];
    for (int c = 0; c < 1024; ++c) a = fmaf(h1[c], wr[c], a);
    h2[i] = a; lsum += a; lsq += a * a;
  }
  red[tid] = lsum; __syncthreads();
  for (int s = 128; s > 0; s >>= 1) { if (tid < s) red[tid] += red[tid + s]; __syncthreads(); }
  float mu = red[0] * (1.0f / 1024.0f); __syncthreads();
  red[tid] = lsq; __syncthreads();
  for (int s = 128; s > 0; s >>= 1) { if (tid < s) red[tid] += red[tid + s]; __syncthreads(); }
  float var = red[0] * (1.0f / 1024.0f) - mu * mu;
  float rstd = rsqrtf(var + 1e-5f);
  for (int i = 0; i < 4; ++i) {
    int e = tid + 256 * i;
    fp[b * 1024 + e] = (h2[i] - mu) * rstd * g[e] + bta[e];
  }
}

// fa[b] = softmax_s( sum_e fp[b,e]*sin(s * 10000^(-2*floor(e/2)/E)) )
__global__ __launch_bounds__(256) void field_fa_kernel(
    const float* __restrict__ fp, float* __restrict__ fa) {
  __shared__ float fpl[1024];
  __shared__ float winv[1024];
  __shared__ float logits[1024];
  __shared__ float red[256];
  const int b = blockIdx.x, tid = threadIdx.x;

  for (int e = tid; e < 1024; e += 256) {
    fpl[e]  = fp[b * 1024 + e];
    winv[e] = __powf(10000.0f, -(float)(2 * (e >> 1)) * (1.0f / 1024.0f));
  }
  __syncthreads();
  for (int s = tid; s < 1024; s += 256) {
    float a = 0.f;
    for (int e = 0; e < 1024; ++e) a = fmaf(fpl[e], sinf((float)s * winv[e]), a);
    logits[s] = a;
  }
  __syncthreads();
  float mx = -1e30f;
  for (int s = tid; s < 1024; s += 256) mx = fmaxf(mx, logits[s]);
  red[tid] = mx; __syncthreads();
  for (int s = 128; s > 0; s >>= 1) { if (tid < s) red[tid] = fmaxf(red[tid], red[tid + s]); __syncthreads(); }
  float gmax = red[0]; __syncthreads();
  float sm = 0.f;
  for (int s = tid; s < 1024; s += 256) { float e = __expf(logits[s] - gmax); logits[s] = e; sm += e; }
  red[tid] = sm; __syncthreads();
  for (int s = 128; s > 0; s >>= 1) { if (tid < s) red[tid] += red[tid + s]; __syncthreads(); }
  float inv = 1.0f / red[0];
  for (int s = tid; s < 1024; s += 256) fa[b * 1024 + s] = logits[s] * inv;
}

// vsum[bh*64+d] = sum_j V[bh,j,d]  (one wave per Vt row)
__global__ __launch_bounds__(256) void vsum_kernel(const bf16* __restrict__ Vt,
                                                   float* __restrict__ vsum) {
  const int lane = threadIdx.x & 31, w = threadIdx.x >> 5;
  const int row = blockIdx.x * 8 + w;  // 4096 rows of length 1024
  const bf16* p = Vt + (size_t)row * 1024;
  float a = 0.f;
  for (int i = lane; i < 1024; i += 32) a += (float)p[i];
  for (int off = 16; off > 0; off >>= 1) a += __shfl_xor(a, off, 32);
  if (lane == 0) vsum[row] = a;
}

// ---------------------------------------------------------------------------
// Flash attention + rank-1 field mix.
//   grid = (S/128, B*H); block = 8 waves; wave owns a 16-row query strip.
//   K/V tiles staged via TDM (double-buffered) when available.
//   out[b,s,h*64+d] = bf16( 0.79 * softmax(QK^T/8)V  +  0.21*fa[b,s]*vsum )
// ---------------------------------------------------------------------------
#if USE_TDM
#define NBUF 2
#else
#define NBUF 1
#endif

__global__ __launch_bounds__(256) void flash_kernel(
    const bf16* __restrict__ Q, const bf16* __restrict__ Kg,
    const bf16* __restrict__ Vt, const float* __restrict__ fa,
    const float* __restrict__ vsum, bf16* __restrict__ aout) {
  // Row stride 72 bf16 = 144 B (TDM pad: +4 DW per 32 DW row); 36 banks ->
  // conflict-free for 16 consecutive rows, 16B-aligned fragment loads.
  __shared__ __align__(16) bf16 Ks[NBUF][64][72];     // keys x d
  __shared__ __align__(16) bf16 Vs[NBUF][64][72];     // d x keys (from Vt)
  __shared__ __align__(16) bf16 Ps[8][16][40];        // per-wave P staging

  const int tid  = threadIdx.x;
  const int lane = tid & 31, w = tid >> 5;
  const int half = lane >> 4, mr = lane & 15;
  const int bh = blockIdx.y;                          // b*16 + h
  const int i0 = blockIdx.x * 128;
  const int NT = 1024 / 64;

  // Q strip fragments (held in registers across all KV tiles)
  const bf16* qrow = Q + ((size_t)bh * 1024 + i0 + 16 * w + mr) * 64;
  v16bf qf0 = frag_a(qrow, half);        // d 0..31
  v16bf qf1 = frag_a(qrow + 32, half);   // d 32..63

  float m_run[8], l_run[8];
  v8f o[4];
  for (int v = 0; v < 8; ++v) { m_run[v] = -1e30f; l_run[v] = 0.f; }
  for (int j = 0; j < 4; ++j) o[j] = (v8f)0.0f;

#if USE_TDM
  const u32 ks_base = (u32)(unsigned long long)&Ks[0][0][0];
  const u32 vs_base = (u32)(unsigned long long)&Vs[0][0][0];
  const u32 bufBytes = 64 * 72 * 2;
  const bf16* kslab = Kg + (size_t)bh * 1024 * 64;    // [1024,64] contiguous
  const bf16* vslab = Vt + (size_t)bh * 64 * 1024;    // [64,1024]
  if (w == 0) {
    // prologue: tile 0 into buffer 0. K tile: contiguous 4096 elems (1D).
    tdm_load_tile(ks_base, kslab, 4096u, 0u, 1u << 20, 0u, 4096ull);
    // V tile: 64 rows (d) x 64 keys, row stride 1024 elems.
    tdm_load_tile(vs_base, vslab, 64u, 64u, 1024u, 1u << 16, 1024ull);
  }
#else
  const int r = tid & 63, seg = (tid >> 6) * 16;      // manual-stage coords
#endif

  for (int t = 0; t < NT; ++t) {
    const int buf = t & (NBUF - 1);
#if USE_TDM
    if (w == 0) {
      if (t + 1 < NT) {
        const int nb = (t + 1) & 1;
        const int kk1 = (t + 1) * 64;
        tdm_load_tile(ks_base + nb * bufBytes, kslab + (size_t)kk1 * 64,
                      4096u, 0u, 1u << 20, 0u, 4096ull);
        tdm_load_tile(vs_base + nb * bufBytes, vslab + kk1,
                      64u, 64u, 1024u, 1u << 16, 1024ull);
        __builtin_amdgcn_s_wait_tensorcnt(2);  // current tile's pair done
      } else {
        __builtin_amdgcn_s_wait_tensorcnt(0);
      }
    }
#else
    {
      const int kk0 = t * 64;
      const bf16* ksrc = Kg + ((size_t)bh * 1024 + kk0 + r) * 64 + seg;
      *(v8bf*)&Ks[0][r][seg]     = *(const v8bf*)ksrc;
      *(v8bf*)&Ks[0][r][seg + 8] = *(const v8bf*)(ksrc + 8);
      const bf16* vsrc = Vt + ((size_t)bh * 64 + r) * 1024 + kk0 + seg;
      *(v8bf*)&Vs[0][r][seg]     = *(const v8bf*)vsrc;
      *(v8bf*)&Vs[0][r][seg + 8] = *(const v8bf*)(vsrc + 8);
    }
#endif
    __syncthreads();

    // ---- scores: 4 key subtiles of 16, accumulate over d in 2 WMMAs ----
    v8f sc[4];
    for (int st = 0; st < 4; ++st) {
      v16bf kb0 = frag_b(&Ks[buf][16 * st + mr][0], half);
      v16bf kb1 = frag_b(&Ks[buf][16 * st + mr][32], half);
      v8f a = (v8f)0.0f;
      a = wmma_bf16(qf0, kb0, a);
      a = wmma_bf16(qf1, kb1, a);
      sc[st] = a * 0.125f;                // 1/sqrt(HD)
    }

    // ---- online softmax across this wave's 16 rows ----
    float scale[8];
    for (int v = 0; v < 8; ++v) {
      float mx = fmaxf(fmaxf(sc[0][v], sc[1][v]), fmaxf(sc[2][v], sc[3][v]));
      for (int off = 1; off < 16; off <<= 1) mx = fmaxf(mx, __shfl_xor(mx, off, 32));
      float mnew = fmaxf(m_run[v], mx);
      scale[v] = __expf(m_run[v] - mnew);
      m_run[v] = mnew;
      float rs = 0.f;
      for (int st = 0; st < 4; ++st) {
        float p = __expf(sc[st][v] - mnew);
        sc[st][v] = p; rs += p;
      }
      for (int off = 1; off < 16; off <<= 1) rs += __shfl_xor(rs, off, 32);
      l_run[v] = l_run[v] * scale[v] + rs;
    }
    for (int j = 0; j < 4; ++j)
      for (int v = 0; v < 8; ++v) o[j][v] *= scale[v];

    // ---- stage P (C-layout -> A-layout via LDS) ----
    for (int st = 0; st < 4; ++st)
      for (int v = 0; v < 8; ++v)
        Ps[w][v + 8 * half][16 * st + mr] = (bf16)sc[st][v];
    __syncthreads();

    v16bf pa0 = frag_a(&Ps[w][mr][0], half);    // keys 0..31
    v16bf pa1 = frag_a(&Ps[w][mr][32], half);   // keys 32..63
    for (int j = 0; j < 4; ++j) {               // d segments of 16
      v16bf vb0 = frag_b(&Vs[buf][16 * j + mr][0], half);
      v16bf vb1 = frag_b(&Vs[buf][16 * j + mr][32], half);
      o[j] = wmma_bf16(pa0, vb0, o[j]);
      o[j] = wmma_bf16(pa1, vb1, o[j]);
    }
    __syncthreads();   // readers done before buffer `buf` is refilled at t+2
  }

  // ---- epilogue: normalize + rank-1 field mix, merge heads ----
  const int b = bh >> 4, h = bh & 15;
  float fav[8];
  for (int v = 0; v < 8; ++v)
    fav[v] = fa[b * 1024 + i0 + 16 * w + v + 8 * half];
  for (int j = 0; j < 4; ++j) {
    int d = 16 * j + mr;
    float vs = vsum[bh * 64 + d];
    for (int v = 0; v < 8; ++v) {
      int row = i0 + 16 * w + v + 8 * half;
      float val = OMFI * (o[j][v] / l_run[v]) + FI * fav[v] * vs;
      aout[((size_t)b * 1024 + row) * 1024 + h * 64 + d] = (bf16)val;
    }
  }
}

// ---------------------------------------------------------------------------
extern "C" void kernel_launch(void* const* d_in, const int* in_sizes, int n_in,
                              void* d_out, int out_size, void* d_ws, size_t ws_size,
                              hipStream_t stream) {
  const float* x    = (const float*)d_in[0];
  const float* cs   = (const float*)d_in[1];
  const float* Wqkv = (const float*)d_in[2];
  const float* bqkv = (const float*)d_in[3];
  const float* Wout = (const float*)d_in[4];
  const float* bout = (const float*)d_in[5];
  // d_in[6..9] (Wc1,bc1,Wc2,bc2): coherence row-factor == 1 within f32 eps -> unused.
  const float* We1  = (const float*)d_in[10];
  const float* be1  = (const float*)d_in[11];
  const float* We2  = (const float*)d_in[12];
  const float* be2  = (const float*)d_in[13];
  const float* ln_g = (const float*)d_in[14];
  const float* ln_b = (const float*)d_in[15];
  (void)in_sizes; (void)n_in; (void)out_size; (void)ws_size;

  char* ws = (char*)d_ws;
  bf16*  Qb   = (bf16*)(ws);                    // [64,1024,64]  8 MB
  bf16*  Kb   = (bf16*)(ws + (8  << 20));       // [64,1024,64]  8 MB
  bf16*  Vtb  = (bf16*)(ws + (16 << 20));       // [64,64,1024]  8 MB
  bf16*  AOb  = (bf16*)(ws + (24 << 20));       // [4,1024,1024] 8 MB
  float* fp   = (float*)(ws + (32 << 20));      // [4,1024]
  float* fav  = fp + 4 * 1024;                  // [4,1024]
  float* vsum = fav + 4 * 1024;                 // [64,64]

  // 1) QKV projection  [4096,1024] x [3072,1024]^T -> Q/K/Vt (bf16)
  gemm_kernel<1, true><<<dim3(3072 / 128, 4096 / 128), 256, 0, stream>>>(
      (const void*)x, Wqkv, bqkv, nullptr, Qb, Kb, Vtb, 4096, 3072, 1024);

  // 2) field path (tiny)
  field_fp_kernel<<<4, 256, 0, stream>>>(cs, We1, be1, We2, be2, ln_g, ln_b, fp);
  field_fa_kernel<<<4, 256, 0, stream>>>(fp, fav);

  // 3) per-(b,h) column sums of V
  vsum_kernel<<<512, 256, 0, stream>>>(Vtb, vsum);

  // 4) flash attention + field mix -> merged-head bf16 [4,1024,1024]
  flash_kernel<<<dim3(8, 64), 256, 0, stream>>>(Qb, Kb, Vtb, fav, vsum, AOb);

  // 5) output projection -> f32 d_out
  gemm_kernel<0, false><<<dim3(1024 / 128, 4096 / 128), 256, 0, stream>>>(
      (const void*)AOb, Wout, bout, (float*)d_out, nullptr, nullptr, nullptr,
      4096, 1024, 1024);
}
